// MGGPT_83485574299872
// MI455X (gfx1250) — matrix operand order repaired and checked by reading, not compiled, not run on hardware
//
#include <hip/hip_runtime.h>
#include <hip/hip_bf16.h>
#include <math.h>

// ---------------------------------------------------------------------------
// MI455X (gfx1250) implementation: GAT x2 -> gated fixup -> GPT2 x2 -> combine
// All dense math on v_wmma_f32_16x16x32_f16 (wave32, 16x16 tiles).
// ---------------------------------------------------------------------------

typedef float    v4f  __attribute__((ext_vector_type(4)));
typedef float    v8f  __attribute__((ext_vector_type(8)));
typedef _Float16 v8h  __attribute__((ext_vector_type(8)));
typedef _Float16 v16h __attribute__((ext_vector_type(16)));

#define NN   32768
#define GG   32
#define TT   1024
#define EE   524288
#define DIN  64
#define DD   128

__device__ inline v8f zero8() {
  v8f z;
#pragma unroll
  for (int i = 0; i < 8; ++i) z[i] = 0.f;
  return z;
}

__device__ inline v8h cvt8(const float* p) {
  v4f a = *(const v4f*)p;
  v4f b = *(const v4f*)(p + 4);
  v8h r;
  r[0] = (_Float16)a[0]; r[1] = (_Float16)a[1]; r[2] = (_Float16)a[2]; r[3] = (_Float16)a[3];
  r[4] = (_Float16)b[0]; r[5] = (_Float16)b[1]; r[6] = (_Float16)b[2]; r[7] = (_Float16)b[3];
  return r;
}

__device__ inline v16h cat16(v8h lo, v8h hi) {
  return __builtin_shufflevector(lo, hi, 0,1,2,3,4,5,6,7,8,9,10,11,12,13,14,15);
}

__device__ inline v8f wmma_f16(v16h a, v16h b, v8f c) {
  // v_wmma_f32_16x16x32_f16 : D = A(16x32 f16) * B(32x16 f16) + C(16x16 f32)
  return __builtin_amdgcn_wmma_f32_16x16x32_f16(false, a, false, b, (short)0, c, false, false);
}

__device__ inline float gelu_new(float x) {
  return 0.5f * x * (1.f + tanhf(0.7978845608028654f * (x + 0.044715f * x * x * x)));
}

// ---------------------------------------------------------------------------
// Generic WMMA GEMM: C[M,Nc] (op)= A[M,K] @ B[K,Nc]  (fp32 in mem, f16 WMMA)
// block = 128 threads (4 waves) -> 64x64 output tile; wave w owns rows w*16..+15
// MODE: 0 plain, 1 +bias, 2 accumulate into C, 3 C = C + AB + bias, 4 gelu(AB+bias)
// ---------------------------------------------------------------------------
template <int MODE>
__global__ __launch_bounds__(128) void gemm_f16w(const float* __restrict__ A,
                                                 const float* __restrict__ B,
                                                 const float* __restrict__ bias,
                                                 float* __restrict__ C,
                                                 int M, int K, int Nc) {
  __shared__ __align__(16) _Float16 As[64][32];
  __shared__ __align__(16) _Float16 Bt[64][32];   // transposed: [col][k]
  const int tid  = threadIdx.x;
  const int w    = tid >> 5;
  const int lane = tid & 31;
  const int ln16 = lane & 15;
  const int hf   = lane >> 4;        // half-wave select
  const int m0   = blockIdx.y * 64;
  const int n0   = blockIdx.x * 64;

  v8f acc[4];
#pragma unroll
  for (int f = 0; f < 4; ++f) acc[f] = zero8();

  for (int k0 = 0; k0 < K; k0 += 32) {
    // stage A (64x32) as f16
#pragma unroll
    for (int j = 0; j < 4; ++j) {
      int q   = tid + 128 * j;          // 0..511 float4 slots
      int row = q >> 3;
      int c4  = (q & 7) * 4;
      v4f av  = *(const v4f*)(A + (size_t)(m0 + row) * K + k0 + c4);
      As[row][c4 + 0] = (_Float16)av[0];
      As[row][c4 + 1] = (_Float16)av[1];
      As[row][c4 + 2] = (_Float16)av[2];
      As[row][c4 + 3] = (_Float16)av[3];
    }
    // stage B (32x64) transposed as f16
#pragma unroll
    for (int j = 0; j < 4; ++j) {
      int q  = tid + 128 * j;
      int kk = q >> 4;                  // 0..31
      int n4 = (q & 15) * 4;
      v4f bv = *(const v4f*)(B + (size_t)(k0 + kk) * Nc + n0 + n4);
      Bt[n4 + 0][kk] = (_Float16)bv[0];
      Bt[n4 + 1][kk] = (_Float16)bv[1];
      Bt[n4 + 2][kk] = (_Float16)bv[2];
      Bt[n4 + 3][kk] = (_Float16)bv[3];
    }
    __syncthreads();

    const int arow = w * 16 + ln16;     // A-frag: lane holds row, K striped per ISA
    v16h a = cat16(*(const v8h*)&As[arow][hf * 8],
                   *(const v8h*)&As[arow][16 + hf * 8]);
#pragma unroll
    for (int f = 0; f < 4; ++f) {
      const int bcol = f * 16 + ln16;   // B-frag: lane holds column, K striped per ISA
      v16h b = cat16(*(const v8h*)&Bt[bcol][hf * 8],
                     *(const v8h*)&Bt[bcol][16 + hf * 8]);
      acc[f] = wmma_f16(a, b, acc[f]);
    }
    __syncthreads();
  }

#pragma unroll
  for (int f = 0; f < 4; ++f) {
    const int col = n0 + f * 16 + ln16;
    float bval = 0.f;
    if (MODE == 1 || MODE == 3 || MODE == 4) bval = bias[col];
#pragma unroll
    for (int r = 0; r < 8; ++r) {
      const int row = m0 + w * 16 + r + hf * 8;  // C layout: VGPR r -> M = r + 8*hf
      const size_t idx = (size_t)row * Nc + col;
      float v = acc[f][r];
      if (MODE == 0)      C[idx] = v;
      else if (MODE == 1) C[idx] = v + bval;
      else if (MODE == 2) C[idx] += v;
      else if (MODE == 3) C[idx] = C[idx] + v + bval;
      else                C[idx] = gelu_new(v + bval);
    }
  }
}

// ---------------------------------------------------------------------------
// GAT helper kernels
// ---------------------------------------------------------------------------
__global__ __launch_bounds__(256) void esed_kernel(const float* __restrict__ hb,
                                                   const float* __restrict__ asr,
                                                   const float* __restrict__ adt,
                                                   float* __restrict__ es,
                                                   float* __restrict__ ed) {
  const int n    = blockIdx.x * 8 + (threadIdx.x >> 5);
  const int lane = threadIdx.x & 31;
  const float* hp = hb + (size_t)n * 256;
  v4f h0 = *(const v4f*)(hp + lane * 4);
  v4f h1 = *(const v4f*)(hp + 128 + lane * 4);
  v4f a0 = *(const v4f*)(asr + lane * 4);
  v4f a1 = *(const v4f*)(asr + 128 + lane * 4);
  v4f d0 = *(const v4f*)(adt + lane * 4);
  v4f d1 = *(const v4f*)(adt + 128 + lane * 4);
  float s0 = 0.f, s1 = 0.f, t0 = 0.f, t1 = 0.f;
#pragma unroll
  for (int i = 0; i < 4; ++i) {
    s0 += h0[i] * a0[i]; s1 += h1[i] * a1[i];
    t0 += h0[i] * d0[i]; t1 += h1[i] * d1[i];
  }
#pragma unroll
  for (int off = 1; off < 32; off <<= 1) {
    s0 += __shfl_xor(s0, off); s1 += __shfl_xor(s1, off);
    t0 += __shfl_xor(t0, off); t1 += __shfl_xor(t1, off);
  }
  if (lane == 0) {
    es[n * 2] = s0; es[n * 2 + 1] = s1;
    ed[n * 2] = t0; ed[n * 2 + 1] = t1;
  }
}

__device__ inline void atomicMaxF(float* a, float v) {
  if (v >= 0.f) atomicMax((int*)a, __float_as_int(v));
  else          atomicMin((unsigned int*)a, __float_as_uint(v));
}

__global__ void edge1_kernel(const int* __restrict__ ei, const float* __restrict__ es,
                             const float* __restrict__ ed, float* __restrict__ eb,
                             float* __restrict__ mb) {
  const int e = blockIdx.x * 256 + threadIdx.x;
  if (e >= EE) return;
  const int s = ei[e], d = ei[EE + e];
#pragma unroll
  for (int h = 0; h < 2; ++h) {
    float v = es[s * 2 + h] + ed[d * 2 + h];
    v = (v > 0.f) ? v : 0.2f * v;           // leaky_relu 0.2
    eb[e * 2 + h] = v;
    atomicMaxF(&mb[d * 2 + h], v);
  }
}

__global__ void mfix_kernel(float* m, int n) {
  const int i = blockIdx.x * 256 + threadIdx.x;
  if (i < n && m[i] < -3.0e38f) m[i] = 0.f;   // no in-edges -> 0 (isfinite fixup)
}

__global__ void edge2_kernel(const int* __restrict__ ei, float* __restrict__ eb,
                             const float* __restrict__ mb, float* __restrict__ dn) {
  const int e = blockIdx.x * 256 + threadIdx.x;
  if (e >= EE) return;
  const int d = ei[EE + e];
#pragma unroll
  for (int h = 0; h < 2; ++h) {
    float ex = __expf(eb[e * 2 + h] - mb[d * 2 + h]);
    eb[e * 2 + h] = ex;
    atomicAdd(&dn[d * 2 + h], ex);
  }
}

// One wave per edge; heads mean-combined (0.5*) so one atomic per (edge, feature)
__global__ __launch_bounds__(256) void edge3_kernel(const int* __restrict__ ei,
                                                    const float* __restrict__ eb,
                                                    const float* __restrict__ dn,
                                                    const float* __restrict__ hb,
                                                    float* __restrict__ o) {
  const int e    = blockIdx.x * 8 + (threadIdx.x >> 5);
  const int lane = threadIdx.x & 31;
  const int s = ei[e], d = ei[EE + e];
  const float a0 = 0.5f * eb[e * 2]     / (dn[d * 2]     + 1e-16f);
  const float a1 = 0.5f * eb[e * 2 + 1] / (dn[d * 2 + 1] + 1e-16f);
  v4f h0 = *(const v4f*)(hb + (size_t)s * 256 + lane * 4);
  v4f h1 = *(const v4f*)(hb + (size_t)s * 256 + 128 + lane * 4);
  float* op = o + (size_t)d * 128 + lane * 4;
#pragma unroll
  for (int i = 0; i < 4; ++i) atomicAdd(op + i, a0 * h0[i] + a1 * h1[i]);
}

// ---------------------------------------------------------------------------
// Elementwise kernels
// ---------------------------------------------------------------------------
__global__ void fillf_kernel(float* p, float v, int n) {
  const int i = blockIdx.x * 256 + threadIdx.x;
  if (i < n) p[i] = v;
}

__global__ void addbias_kernel(float* p, const float* b) {
  const int i = blockIdx.x * 256 + threadIdx.x;
  p[i] += b[i & 127];
}

__global__ void fixup_kernel(float* __restrict__ xg, const float* __restrict__ t,
                             const float* __restrict__ hN, const float* __restrict__ bb1,
                             const float* __restrict__ bb2) {
  const int i = blockIdx.x * 256 + threadIdx.x;
  const int d = i & 127;
  const float beta = tanhf(t[i] + bb1[d] + bb2[d]);
  xg[i] = xg[i] + beta - hN[i];
}

__global__ void embed_kernel(float* __restrict__ hg, const float* __restrict__ xg,
                             const float* __restrict__ wpe) {
  const int i = blockIdx.x * 256 + threadIdx.x;
  hg[i] = xg[i] + wpe[i & (TT * DD - 1)];   // pos = node % 1024
}

// LayerNorm: one wave per 128-wide row
__global__ __launch_bounds__(128) void ln_kernel(float* __restrict__ out,
                                                 const float* __restrict__ in,
                                                 const float* __restrict__ g,
                                                 const float* __restrict__ b) {
  const int row  = blockIdx.x * 4 + (threadIdx.x >> 5);
  const int lane = threadIdx.x & 31;
  v4f x = *(const v4f*)(in + (size_t)row * 128 + lane * 4);
  float s = x[0] + x[1] + x[2] + x[3];
#pragma unroll
  for (int off = 1; off < 32; off <<= 1) s += __shfl_xor(s, off);
  const float mu = s * (1.f / 128.f);
  v4f xc; float vs = 0.f;
#pragma unroll
  for (int i = 0; i < 4; ++i) { xc[i] = x[i] - mu; vs += xc[i] * xc[i]; }
#pragma unroll
  for (int off = 1; off < 32; off <<= 1) vs += __shfl_xor(vs, off);
  const float rs = rsqrtf(vs * (1.f / 128.f) + 1e-5f);
  v4f o;
#pragma unroll
  for (int i = 0; i < 4; ++i) o[i] = xc[i] * rs * g[lane * 4 + i] + b[lane * 4 + i];
  *(v4f*)(out + (size_t)row * 128 + lane * 4) = o;
}

// ---------------------------------------------------------------------------
// Causal flash-attention: one wave per (graph, head, 16-row q tile).
// T=1024, HD=64, scale 1/8. Online softmax; P transposed through per-wave LDS.
// ---------------------------------------------------------------------------
__global__ __launch_bounds__(128) void attn_kernel(const float* __restrict__ qkv,
                                                   float* __restrict__ out) {
  __shared__ __align__(16) _Float16 pbuf[4][16][32];
  const int w    = threadIdx.x >> 5;
  const int lane = threadIdx.x & 31;
  const int ln16 = lane & 15;
  const int hf   = lane >> 4;
  const int wid  = blockIdx.x * 4 + w;
  const int g    = wid >> 7;
  const int rem  = wid & 127;
  const int h    = rem >> 6;
  const int qt   = rem & 63;

  const float* base = qkv + (size_t)g * TT * 384;
  const float* kp_b = base + 128 + h * 64;
  const float* vp_b = base + 256 + h * 64;

  // Q fragments (16 rows x 64 feats -> two A-frags)
  const int qrow = qt * 16 + ln16;
  const float* qp = base + (size_t)qrow * 384 + h * 64;
  v16h aq0 = cat16(cvt8(qp + hf * 16),      cvt8(qp + 8 + hf * 16));
  v16h aq1 = cat16(cvt8(qp + 32 + hf * 16), cvt8(qp + 40 + hf * 16));

  float m_[8], l_[8];
  v8f oacc[4];
#pragma unroll
  for (int r = 0; r < 8; ++r) { m_[r] = -1e30f; l_[r] = 0.f; }
#pragma unroll
  for (int f = 0; f < 4; ++f) oacc[f] = zero8();

  const int kbmax = (qt * 16 + 15) >> 5;
  for (int kb = 0; kb <= kbmax; ++kb) {
    const int kb0 = kb * 32;
    v8f c0 = zero8(), c1 = zero8();
    {
      const float* kp0 = kp_b + (size_t)(kb0 + ln16) * 384 + hf * 16;
      const float* kp1 = kp_b + (size_t)(kb0 + 16 + ln16) * 384 + hf * 16;
      v16h b0 = cat16(cvt8(kp0), cvt8(kp0 + 8));
      v16h b1 = cat16(cvt8(kp1), cvt8(kp1 + 8));
      c0 = wmma_f16(aq0, b0, c0);
      c1 = wmma_f16(aq0, b1, c1);
      b0 = cat16(cvt8(kp0 + 32), cvt8(kp0 + 40));
      b1 = cat16(cvt8(kp1 + 32), cvt8(kp1 + 40));
      c0 = wmma_f16(aq1, b0, c0);
      c1 = wmma_f16(aq1, b1, c1);
    }
    // scale, causal mask, online softmax update (half-wave row reductions)
#pragma unroll
    for (int r = 0; r < 8; ++r) {
      const int rowt = qt * 16 + r + hf * 8;
      float v0 = c0[r] * 0.125f; if (kb0 + ln16 > rowt)      v0 = -1e30f;
      float v1 = c1[r] * 0.125f; if (kb0 + 16 + ln16 > rowt) v1 = -1e30f;
      float mx = fmaxf(v0, v1);
      mx = fmaxf(mx, __shfl_xor(mx, 1));
      mx = fmaxf(mx, __shfl_xor(mx, 2));
      mx = fmaxf(mx, __shfl_xor(mx, 4));
      mx = fmaxf(mx, __shfl_xor(mx, 8));
      const float mn = fmaxf(m_[r], mx);
      const float al = __expf(m_[r] - mn);
      m_[r] = mn;
      v0 = __expf(v0 - mn);
      v1 = __expf(v1 - mn);
      float sm = v0 + v1;
      sm += __shfl_xor(sm, 1); sm += __shfl_xor(sm, 2);
      sm += __shfl_xor(sm, 4); sm += __shfl_xor(sm, 8);
      l_[r] = l_[r] * al + sm;
      oacc[0][r] *= al; oacc[1][r] *= al; oacc[2][r] *= al; oacc[3][r] *= al;
      pbuf[w][r + hf * 8][ln16]      = (_Float16)v0;
      pbuf[w][r + hf * 8][16 + ln16] = (_Float16)v1;
    }
    // wave-local LDS transpose of P (C layout -> A fragment layout)
    asm volatile("s_wait_dscnt 0" ::: "memory");
    v16h ap = cat16(*(const v8h*)&pbuf[w][ln16][hf * 8],
                    *(const v8h*)&pbuf[w][ln16][16 + hf * 8]);
#pragma unroll
    for (int f = 0; f < 4; ++f) {
      const int feat = f * 16 + ln16;
      v16h bv;
#pragma unroll
      for (int i = 0; i < 16; ++i) {
        const int key = kb0 + (i < 8 ? hf * 8 + i : 16 + hf * 8 + (i - 8));
        bv[i] = (_Float16)vp_b[(size_t)key * 384 + feat];
      }
      oacc[f] = wmma_f16(ap, bv, oacc[f]);
    }
  }
#pragma unroll
  for (int f = 0; f < 4; ++f) {
#pragma unroll
    for (int r = 0; r < 8; ++r) {
      const int rowt = qt * 16 + r + hf * 8;
      out[(size_t)(g * TT + rowt) * 128 + h * 64 + f * 16 + ln16] = oacc[f][r] / l_[r];
    }
  }
}

// ---------------------------------------------------------------------------
// Host orchestration
// ---------------------------------------------------------------------------
extern "C" void kernel_launch(void* const* d_in, const int* in_sizes, int n_in,
                              void* d_out, int out_size, void* d_ws, size_t ws_size,
                              hipStream_t stream) {
  (void)in_sizes; (void)n_in; (void)out_size; (void)ws_size;
  const float* x    = (const float*)d_in[0];
  const int*   ei   = (const int*)d_in[1];
  const float* W1   = (const float*)d_in[3];
  const float* as1  = (const float*)d_in[4];
  const float* ad1  = (const float*)d_in[5];
  const float* b1   = (const float*)d_in[6];
  const float* W2   = (const float*)d_in[7];
  const float* as2  = (const float*)d_in[8];
  const float* ad2  = (const float*)d_in[9];
  const float* b2   = (const float*)d_in[10];
  const float* Wb1  = (const float*)d_in[15];
  const float* bb1  = (const float*)d_in[16];
  const float* Wb2  = (const float*)d_in[17];
  const float* bb2  = (const float*)d_in[18];
  const float* wpe  = (const float*)d_in[19];
  const float* ln1g = (const float*)d_in[20];
  const float* ln1b = (const float*)d_in[21];
  const float* qkvw = (const float*)d_in[22];
  const float* qkvb = (const float*)d_in[23];
  const float* pw   = (const float*)d_in[24];
  const float* pb   = (const float*)d_in[25];
  const float* ln2g = (const float*)d_in[26];
  const float* ln2b = (const float*)d_in[27];
  const float* fcw  = (const float*)d_in[28];
  const float* fcb  = (const float*)d_in[29];
  const float* fc2w = (const float*)d_in[30];
  const float* fc2b = (const float*)d_in[31];
  const float* lnfg = (const float*)d_in[32];
  const float* lnfb = (const float*)d_in[33];
  const float* Wc   = (const float*)d_in[34];
  const float* bc   = (const float*)d_in[35];
  float* out = (float*)d_out;

  // workspace layout (floats); "big" is reused: GAT h [N,256] / qkv [N,384] / mlp [N,512]
  float* ws  = (float*)d_ws;
  float* big = ws;
  float* xg  = big + (size_t)NN * 512;
  float* hg  = xg  + (size_t)NN * 128;      // h_N_v, later GPT hidden state
  float* tmp = hg  + (size_t)NN * 128;      // beta pre-act / LN out / attn out / x_gpt
  float* eb  = tmp + (size_t)NN * 128;      // edge buffer [E,2]
  float* es  = eb  + (size_t)EE * 2;
  float* ed  = es  + NN * 2;
  float* mb  = ed  + NN * 2;
  float* dn  = mb  + NN * 2;

  const dim3 b128(128), b256(256);
  const float ninf = -__builtin_huge_valf();
  const int eBlocks = EE / 256, nePerWave = EE / 8;
  const int ndBlocks = (NN * 128) / 256, n2Blocks = (NN * 2) / 256;

  // ---------------- GAT layer 1 ----------------
  gemm_f16w<0><<<dim3(4, NN / 64), b128, 0, stream>>>(x, W1, nullptr, big, NN, DIN, 256);
  esed_kernel<<<NN / 8, b256, 0, stream>>>(big, as1, ad1, es, ed);
  fillf_kernel<<<n2Blocks, b256, 0, stream>>>(mb, ninf, NN * 2);
  fillf_kernel<<<n2Blocks, b256, 0, stream>>>(dn, 0.f, NN * 2);
  fillf_kernel<<<ndBlocks, b256, 0, stream>>>(xg, 0.f, NN * 128);
  edge1_kernel<<<eBlocks, b256, 0, stream>>>(ei, es, ed, eb, mb);
  mfix_kernel<<<n2Blocks, b256, 0, stream>>>(mb, NN * 2);
  edge2_kernel<<<eBlocks, b256, 0, stream>>>(ei, eb, mb, dn);
  edge3_kernel<<<nePerWave, b256, 0, stream>>>(ei, eb, dn, big, xg);
  addbias_kernel<<<ndBlocks, b256, 0, stream>>>(xg, b1);

  // ---------------- GAT layer 2 ----------------
  gemm_f16w<0><<<dim3(4, NN / 64), b128, 0, stream>>>(xg, W2, nullptr, big, NN, 128, 256);
  esed_kernel<<<NN / 8, b256, 0, stream>>>(big, as2, ad2, es, ed);
  fillf_kernel<<<n2Blocks, b256, 0, stream>>>(mb, ninf, NN * 2);
  fillf_kernel<<<n2Blocks, b256, 0, stream>>>(dn, 0.f, NN * 2);
  fillf_kernel<<<ndBlocks, b256, 0, stream>>>(hg, 0.f, NN * 128);
  edge1_kernel<<<eBlocks, b256, 0, stream>>>(ei, es, ed, eb, mb);
  mfix_kernel<<<n2Blocks, b256, 0, stream>>>(mb, NN * 2);
  edge2_kernel<<<eBlocks, b256, 0, stream>>>(ei, eb, mb, dn);
  edge3_kernel<<<nePerWave, b256, 0, stream>>>(ei, eb, dn, big, hg);
  addbias_kernel<<<ndBlocks, b256, 0, stream>>>(hg, b2);

  // ---------------- gated missing-info fixup ----------------
  gemm_f16w<0><<<dim3(2, NN / 64), b128, 0, stream>>>(xg, Wb1, nullptr, tmp, NN, 128, 128);
  gemm_f16w<2><<<dim3(2, NN / 64), b128, 0, stream>>>(hg, Wb2, nullptr, tmp, NN, 128, 128);
  fixup_kernel<<<ndBlocks, b256, 0, stream>>>(xg, tmp, hg, bb1, bb2);

  // ---------------- GPT2 (dense-pad is a pure reshape: batch sorted, T=1024) ----
  embed_kernel<<<ndBlocks, b256, 0, stream>>>(hg, xg, wpe);
  for (int l = 0; l < 2; ++l) {
    ln_kernel<<<NN / 4, b128, 0, stream>>>(tmp, hg, ln1g + l * 128, ln1b + l * 128);
    gemm_f16w<1><<<dim3(6, NN / 64), b128, 0, stream>>>(tmp, qkvw + (size_t)l * 128 * 384,
                                                        qkvb + l * 384, big, NN, 128, 384);
    attn_kernel<<<(GG * 2 * 64) / 4, b128, 0, stream>>>(big, tmp);
    gemm_f16w<3><<<dim3(2, NN / 64), b128, 0, stream>>>(tmp, pw + (size_t)l * 128 * 128,
                                                        pb + l * 128, hg, NN, 128, 128);
    ln_kernel<<<NN / 4, b128, 0, stream>>>(tmp, hg, ln2g + l * 128, ln2b + l * 128);
    gemm_f16w<4><<<dim3(8, NN / 64), b128, 0, stream>>>(tmp, fcw + (size_t)l * 128 * 512,
                                                        fcb + l * 512, big, NN, 128, 512);
    gemm_f16w<3><<<dim3(2, NN / 64), b128, 0, stream>>>(big, fc2w + (size_t)l * 512 * 128,
                                                        fc2b + l * 128, hg, NN, 512, 128);
  }
  ln_kernel<<<NN / 4, b128, 0, stream>>>(tmp, hg, lnfg, lnfb);

  // ---------------- combine: out = [x_gnn, x_gpt] @ Wc + bc ----------------
  gemm_f16w<1><<<dim3(2, NN / 64), b128, 0, stream>>>(xg, Wc, bc, out, NN, 128, 128);
  gemm_f16w<2><<<dim3(2, NN / 64), b128, 0, stream>>>(tmp, Wc + 128 * 128, nullptr, out, NN, 128, 128);
  // second tuple output: zeros
  fillf_kernel<<<ndBlocks, b256, 0, stream>>>(out + (size_t)NN * 128, 0.f, NN * 128);
}
